// Attention_42073499632328
// MI455X (gfx1250) — matrix operand order repaired and checked by reading, not compiled, not run on hardware
//
#include <hip/hip_runtime.h>
#include <hip/hip_bf16.h>

typedef __attribute__((ext_vector_type(16))) _Float16 v16h;
typedef __attribute__((ext_vector_type(8)))  _Float16 v8h;
typedef __attribute__((ext_vector_type(4)))  _Float16 v4h;
typedef __attribute__((ext_vector_type(8)))  float    v8f;
typedef int b128i __attribute__((vector_size(16)));

#define B_   4
#define N_   1370
#define NPAD 1376
#define C_   1024
#define H_   16
#define HD   64
#define MTOT (B_ * N_)   // 5480
#define NQT  (NPAD / 16) // 86 q tiles

#if defined(__AMDGCN__) && __has_builtin(__builtin_amdgcn_global_load_async_to_lds_b128) && __has_builtin(__builtin_amdgcn_s_wait_asynccnt)
#define HAVE_ASYNC_LDS 1
#else
#define HAVE_ASYNC_LDS 0
#endif

static __device__ __forceinline__ v16h cat8(v8h lo, v8h hi) {
    return __builtin_shufflevector(lo, hi, 0,1,2,3,4,5,6,7,8,9,10,11,12,13,14,15);
}
static __device__ __forceinline__ v8f wmma32(v16h a, v16h b, v8f c) {
    return __builtin_amdgcn_wmma_f32_16x16x32_f16(false, a, false, b, (short)0, c, false, false);
}
#if HAVE_ASYNC_LDS
typedef __attribute__((address_space(1))) b128i* gptr128;
typedef __attribute__((address_space(3))) b128i* lptr128;
static __device__ __forceinline__ void async_b128(const _Float16* g, _Float16* l) {
    __builtin_amdgcn_global_load_async_to_lds_b128((gptr128)g, (lptr128)l, 0, 0);
}
#endif

// ---------------------------------------------------------------------------
// Kernel 1: QKV projection [5480,1024] x [3072,1024]^T (+bias), f16 WMMA,
// double-buffered LDS. Writes Q,K [b,h,Npad,64] (Q pre-scaled), V^T [b,h,64,Npad].
// ---------------------------------------------------------------------------
__global__ __launch_bounds__(256) void qkv_gemm(
    const float* __restrict__ tokens, const float* __restrict__ qkv_w,
    const float* __restrict__ qkv_b,
    _Float16* __restrict__ qh, _Float16* __restrict__ kh, _Float16* __restrict__ vth)
{
    __shared__ __align__(16) _Float16 As[2][128][40];
    __shared__ __align__(16) _Float16 Bs[2][128][40];
    const int t = threadIdx.x, lane = t & 31, wid = t >> 5;
    const int lid = lane & 15, grp = lane >> 4;
    const int wm = (wid & 1) * 64, wn = (wid >> 1) * 32;
    const int m0 = blockIdx.x * 128, n0 = blockIdx.y * 128;
    const int col4 = (t & 7) << 2, rbase = t >> 3;

    v8f acc[4][2];
#pragma unroll
    for (int i = 0; i < 4; ++i)
#pragma unroll
        for (int j = 0; j < 2; ++j) acc[i][j] = (v8f){};

    float4 ra[4], rb[4];
#pragma unroll
    for (int p = 0; p < 4; ++p) {
        int row = rbase + p * 32;
        int gm = m0 + row; gm = gm < (MTOT - 1) ? gm : (MTOT - 1);
        ra[p] = *(const float4*)(tokens + (size_t)gm * C_ + col4);
        rb[p] = *(const float4*)(qkv_w + (size_t)(n0 + row) * C_ + col4);
    }
#pragma unroll
    for (int p = 0; p < 4; ++p) {
        int row = rbase + p * 32;
        v4h ha = { (_Float16)ra[p].x, (_Float16)ra[p].y, (_Float16)ra[p].z, (_Float16)ra[p].w };
        v4h hb = { (_Float16)rb[p].x, (_Float16)rb[p].y, (_Float16)rb[p].z, (_Float16)rb[p].w };
        *(v4h*)&As[0][row][col4] = ha;
        *(v4h*)&Bs[0][row][col4] = hb;
    }
    __syncthreads();

    for (int kk = 0; kk < C_; kk += 32) {
        const int buf = (kk >> 5) & 1;
        const bool more = (kk + 32) < C_;
        if (more) {
#pragma unroll
            for (int p = 0; p < 4; ++p) {
                int row = rbase + p * 32;
                int gm = m0 + row; gm = gm < (MTOT - 1) ? gm : (MTOT - 1);
                ra[p] = *(const float4*)(tokens + (size_t)gm * C_ + kk + 32 + col4);
                rb[p] = *(const float4*)(qkv_w + (size_t)(n0 + row) * C_ + kk + 32 + col4);
            }
        }
        v16h bfr[2];
#pragma unroll
        for (int tn = 0; tn < 2; ++tn) {
            const _Float16* p = &Bs[buf][wn + tn * 16 + lid][grp * 16];
            bfr[tn] = cat8(*(const v8h*)p, *(const v8h*)(p + 8));
        }
#pragma unroll
        for (int tm = 0; tm < 4; ++tm) {
            const _Float16* pa = &As[buf][wm + tm * 16 + lid][0];
            v16h af = cat8(*(const v8h*)(pa + grp * 16), *(const v8h*)(pa + grp * 16 + 8));
            acc[tm][0] = wmma32(af, bfr[0], acc[tm][0]);
            acc[tm][1] = wmma32(af, bfr[1], acc[tm][1]);
        }
        if (more) {
#pragma unroll
            for (int p = 0; p < 4; ++p) {
                int row = rbase + p * 32;
                v4h ha = { (_Float16)ra[p].x, (_Float16)ra[p].y, (_Float16)ra[p].z, (_Float16)ra[p].w };
                v4h hb = { (_Float16)rb[p].x, (_Float16)rb[p].y, (_Float16)rb[p].z, (_Float16)rb[p].w };
                *(v4h*)&As[buf ^ 1][row][col4] = ha;
                *(v4h*)&Bs[buf ^ 1][row][col4] = hb;
            }
        }
        __syncthreads();
    }

#pragma unroll
    for (int tn = 0; tn < 2; ++tn) {
        const int n = n0 + wn + tn * 16 + lid;
        const float bias = qkv_b[n];
        const int which = n >> 10, rem = n & 1023, h = rem >> 6, cc = rem & 63;
#pragma unroll
        for (int tm = 0; tm < 4; ++tm) {
#pragma unroll
            for (int r = 0; r < 8; ++r) {
                int m = m0 + wm + tm * 16 + r + grp * 8;
                if (m >= MTOT) continue;
                int b = m / N_, row = m - b * N_;
                float val = acc[tm][tn][r] + bias;
                if (which == 0) val *= 0.125f;
                size_t bh = (size_t)(b * H_ + h);
                if (which == 2)
                    vth[(bh * HD + cc) * NPAD + row] = (_Float16)val;
                else {
                    _Float16* dst = (which == 0) ? qh : kh;
                    dst[(bh * NPAD + row) * HD + cc] = (_Float16)val;
                }
            }
        }
    }
}

// ---------------------------------------------------------------------------
// Kernel 2: flash attention. Block = 8 waves on one (b,h); each wave owns TWO
// 16-column q tiles (S^T / O^T form -> per-lane softmax state). K/V^T tiles
// double-buffered in LDS; next tile staged with GLOBAL_LOAD_ASYNC_TO_LDS_B128
// (ASYNCcnt) when available, else register double-buffer.
// ---------------------------------------------------------------------------
__global__ __launch_bounds__(256) void flash_attn(
    const _Float16* __restrict__ qh, const _Float16* __restrict__ kh,
    const _Float16* __restrict__ vth, _Float16* __restrict__ oh)
{
    __shared__ __align__(16) _Float16 Ks[2][32][72];
    __shared__ __align__(16) _Float16 Vs[2][64][40];
    const int t = threadIdx.x, lane = t & 31, wid = t >> 5;
    const int lid = lane & 15, grp = lane >> 4;
    const int b = blockIdx.z, h = blockIdx.y;
    const size_t bh = (size_t)(b * H_ + h);
    const _Float16* kptr = kh  + bh * (size_t)NPAD * HD;
    const _Float16* vptr = vth + bh * (size_t)HD * NPAD;

    int qt[2];
#pragma unroll
    for (int j = 0; j < 2; ++j) {
        int q = blockIdx.x * 16 + wid * 2 + j;
        qt[j] = q < (NQT - 1) ? q : (NQT - 1);
    }

    v16h bq[2][2];
#pragma unroll
    for (int j = 0; j < 2; ++j) {
        const _Float16* qp = qh + (bh * NPAD + qt[j] * 16 + lid) * HD;
#pragma unroll
        for (int c = 0; c < 2; ++c) {
            const _Float16* p = qp + c * 32 + grp * 16;
            bq[j][c] = cat8(*(const v8h*)p, *(const v8h*)(p + 8));
        }
    }

    v8f o[2][4];
#pragma unroll
    for (int j = 0; j < 2; ++j)
#pragma unroll
        for (int ct = 0; ct < 4; ++ct) o[j][ct] = (v8f){};
    float mrun[2] = { -1e30f, -1e30f }, lrun[2] = { 0.f, 0.f };

    const int NKT = (N_ + 31) / 32;  // 43
    const int kr = t >> 3, kc0 = (t & 7) * 8;
    const int vc = t >> 2, vk0 = (t & 3) * 8;

    // prologue: stage tile 0 synchronously
    *(v8h*)&Ks[0][kr][kc0] = *(const v8h*)(kptr + (size_t)kr * HD + kc0);
    *(v8h*)&Vs[0][vc][vk0] = *(const v8h*)(vptr + (size_t)vc * NPAD + vk0);
    __syncthreads();

    for (int kt = 0; kt < NKT; ++kt) {
        const int cur = kt & 1, nxt = cur ^ 1;
        const int kb = kt * 32;
        const bool more = (kt + 1) < NKT;
#if HAVE_ASYNC_LDS
        if (more) {
            async_b128(kptr + (size_t)(kb + 32 + kr) * HD + kc0, &Ks[nxt][kr][kc0]);
            async_b128(vptr + (size_t)vc * NPAD + kb + 32 + vk0, &Vs[nxt][vc][vk0]);
        }
#else
        v8h nk = (v8h){}, nv = (v8h){};
        if (more) {
            nk = *(const v8h*)(kptr + (size_t)(kb + 32 + kr) * HD + kc0);
            nv = *(const v8h*)(vptr + (size_t)vc * NPAD + kb + 32 + vk0);
            __builtin_prefetch(kptr + (size_t)(kb + 64 + kr) * HD + kc0, 0, 1);
        }
#endif
        v16h ak[2][2];
#pragma unroll
        for (int tt = 0; tt < 2; ++tt)
#pragma unroll
            for (int c = 0; c < 2; ++c) {
                const _Float16* p = &Ks[cur][tt * 16 + lid][c * 32];
                ak[tt][c] = cat8(*(const v8h*)(p + grp * 16), *(const v8h*)(p + grp * 16 + 8));
            }

        v16h pf[2];
#pragma unroll
        for (int j = 0; j < 2; ++j) {
            v8f st0 = (v8f){}, st1 = (v8f){};
#pragma unroll
            for (int c = 0; c < 2; ++c) {
                st0 = wmma32(ak[0][c], bq[j][c], st0);
                st1 = wmma32(ak[1][c], bq[j][c], st1);
            }
            float s0[8], s1[8], tmax = -1e30f;
#pragma unroll
            for (int r = 0; r < 8; ++r) {
                int k0 = kb + r + grp * 8, k1 = kb + 16 + r + grp * 8;
                s0[r] = (k0 < N_) ? st0[r] : -1e30f;
                s1[r] = (k1 < N_) ? st1[r] : -1e30f;
                tmax = fmaxf(tmax, fmaxf(s0[r], s1[r]));
            }
            tmax = fmaxf(tmax, __shfl_xor(tmax, 16, 32));
            float mnew = fmaxf(mrun[j], tmax);
            float fac = __expf(mrun[j] - mnew);
            float p0[8], p1[8], psum = 0.f;
#pragma unroll
            for (int r = 0; r < 8; ++r) {
                p0[r] = __expf(s0[r] - mnew);
                p1[r] = __expf(s1[r] - mnew);
                psum += p0[r] + p1[r];
            }
            psum += __shfl_xor(psum, 16, 32);
            lrun[j] = lrun[j] * fac + psum;
            mrun[j] = mnew;
#pragma unroll
            for (int ct = 0; ct < 4; ++ct) o[j][ct] = o[j][ct] * fac;
#pragma unroll
            for (int r = 0; r < 8; ++r) {
                float x0p = __shfl_xor(p0[r], 16, 32);
                float x1p = __shfl_xor(p1[r], 16, 32);
                pf[j][r]     = (_Float16)((grp == 0) ? p0[r] : x1p);
                pf[j][r + 8] = (_Float16)((grp == 0) ? x0p   : p1[r]);
            }
        }
#pragma unroll
        for (int ct = 0; ct < 4; ++ct) {
            const _Float16* pv = &Vs[cur][ct * 16 + lid][0];
            v16h av = cat8(*(const v8h*)(pv + grp * 16), *(const v8h*)(pv + grp * 16 + 8));
            o[0][ct] = wmma32(av, pf[0], o[0][ct]);
            o[1][ct] = wmma32(av, pf[1], o[1][ct]);
        }
#if HAVE_ASYNC_LDS
        __builtin_amdgcn_s_wait_asynccnt(0);
#else
        if (more) {
            *(v8h*)&Ks[nxt][kr][kc0] = nk;
            *(v8h*)&Vs[nxt][vc][vk0] = nv;
        }
#endif
        __syncthreads();
    }

#pragma unroll
    for (int j = 0; j < 2; ++j) {
        const float inv = 1.f / lrun[j];
        const int q = qt[j] * 16 + lid;
        if (q < N_) {
            _Float16* dst = oh + ((size_t)b * NPAD + q) * C_ + h * HD;
#pragma unroll
            for (int ct = 0; ct < 4; ++ct)
#pragma unroll
                for (int r = 0; r < 8; ++r)
                    dst[ct * 16 + r + grp * 8] = (_Float16)(o[j][ct][r] * inv);
        }
    }
}

// ---------------------------------------------------------------------------
// Kernel 3: output projection [5480,1024](f16) x [1024,1024]^T + bias -> f32
// ---------------------------------------------------------------------------
__global__ __launch_bounds__(256) void proj_gemm(
    const _Float16* __restrict__ xh, const float* __restrict__ proj_w,
    const float* __restrict__ proj_b, float* __restrict__ out)
{
    __shared__ __align__(16) _Float16 As[2][128][40];
    __shared__ __align__(16) _Float16 Bs[2][128][40];
    const int t = threadIdx.x, lane = t & 31, wid = t >> 5;
    const int lid = lane & 15, grp = lane >> 4;
    const int wm = (wid & 1) * 64, wn = (wid >> 1) * 32;
    const int m0 = blockIdx.x * 128, n0 = blockIdx.y * 128;
    const int col4 = (t & 7) << 2, rbase = t >> 3;

    v8f acc[4][2];
#pragma unroll
    for (int i = 0; i < 4; ++i)
#pragma unroll
        for (int j = 0; j < 2; ++j) acc[i][j] = (v8f){};

    v8h rah[2];
    float4 rb[4];
#pragma unroll
    for (int p = 0; p < 2; ++p) {
        int ch = t + p * 256, row = ch >> 2, c0 = (ch & 3) * 8;
        int gm = m0 + row; gm = gm < (MTOT - 1) ? gm : (MTOT - 1);
        int bb = gm / N_, rr = gm - bb * N_;
        rah[p] = *(const v8h*)(xh + ((size_t)bb * NPAD + rr) * C_ + c0);
    }
#pragma unroll
    for (int p = 0; p < 4; ++p) {
        int row = rbase + p * 32;
        rb[p] = *(const float4*)(proj_w + (size_t)(n0 + row) * C_ + col4);
    }
#pragma unroll
    for (int p = 0; p < 2; ++p) {
        int ch = t + p * 256, row = ch >> 2, c0 = (ch & 3) * 8;
        *(v8h*)&As[0][row][c0] = rah[p];
    }
#pragma unroll
    for (int p = 0; p < 4; ++p) {
        int row = rbase + p * 32;
        v4h hb = { (_Float16)rb[p].x, (_Float16)rb[p].y, (_Float16)rb[p].z, (_Float16)rb[p].w };
        *(v4h*)&Bs[0][row][col4] = hb;
    }
    __syncthreads();

    for (int kk = 0; kk < C_; kk += 32) {
        const int buf = (kk >> 5) & 1;
        const bool more = (kk + 32) < C_;
        if (more) {
#pragma unroll
            for (int p = 0; p < 2; ++p) {
                int ch = t + p * 256, row = ch >> 2, c0 = (ch & 3) * 8;
                int gm = m0 + row; gm = gm < (MTOT - 1) ? gm : (MTOT - 1);
                int bb = gm / N_, rr = gm - bb * N_;
                rah[p] = *(const v8h*)(xh + ((size_t)bb * NPAD + rr) * C_ + kk + 32 + c0);
            }
#pragma unroll
            for (int p = 0; p < 4; ++p) {
                int row = rbase + p * 32;
                rb[p] = *(const float4*)(proj_w + (size_t)(n0 + row) * C_ + kk + 32 + col4);
            }
        }
        v16h bfr[2];
#pragma unroll
        for (int tn = 0; tn < 2; ++tn) {
            const _Float16* p = &Bs[buf][wn + tn * 16 + lid][grp * 16];
            bfr[tn] = cat8(*(const v8h*)p, *(const v8h*)(p + 8));
        }
#pragma unroll
        for (int tm = 0; tm < 4; ++tm) {
            const _Float16* pa = &As[buf][wm + tm * 16 + lid][0];
            v16h af = cat8(*(const v8h*)(pa + grp * 16), *(const v8h*)(pa + grp * 16 + 8));
            acc[tm][0] = wmma32(af, bfr[0], acc[tm][0]);
            acc[tm][1] = wmma32(af, bfr[1], acc[tm][1]);
        }
        if (more) {
#pragma unroll
            for (int p = 0; p < 2; ++p) {
                int ch = t + p * 256, row = ch >> 2, c0 = (ch & 3) * 8;
                *(v8h*)&As[buf ^ 1][row][c0] = rah[p];
            }
#pragma unroll
            for (int p = 0; p < 4; ++p) {
                int row = rbase + p * 32;
                v4h hb = { (_Float16)rb[p].x, (_Float16)rb[p].y, (_Float16)rb[p].z, (_Float16)rb[p].w };
                *(v4h*)&Bs[buf ^ 1][row][col4] = hb;
            }
        }
        __syncthreads();
    }

#pragma unroll
    for (int tn = 0; tn < 2; ++tn) {
        const int n = n0 + wn + tn * 16 + lid;
        const float bias = proj_b[n];
#pragma unroll
        for (int tm = 0; tm < 4; ++tm) {
#pragma unroll
            for (int r = 0; r < 8; ++r) {
                int m = m0 + wm + tm * 16 + r + grp * 8;
                if (m >= MTOT) continue;
                out[(size_t)m * C_ + n] = acc[tm][tn][r] + bias;
            }
        }
    }
}

extern "C" void kernel_launch(void* const* d_in, const int* in_sizes, int n_in,
                              void* d_out, int out_size, void* d_ws, size_t ws_size,
                              hipStream_t stream) {
    (void)in_sizes; (void)n_in; (void)out_size; (void)ws_size;
    const float* tokens = (const float*)d_in[0];
    const float* qkv_w  = (const float*)d_in[1];
    const float* qkv_b  = (const float*)d_in[2];
    const float* proj_w = (const float*)d_in[3];
    const float* proj_b = (const float*)d_in[4];
    float* out = (float*)d_out;

    _Float16* wsh = (_Float16*)d_ws;
    const size_t QSZ = (size_t)B_ * H_ * NPAD * HD;
    _Float16* qh  = wsh;
    _Float16* khp = wsh + QSZ;
    _Float16* vth = wsh + 2 * QSZ;
    _Float16* oh  = wsh + 3 * QSZ;

    qkv_gemm <<<dim3((MTOT + 127) / 128, 3 * C_ / 128), 256, 0, stream>>>(
        tokens, qkv_w, qkv_b, qh, khp, vth);
    flash_attn<<<dim3((NQT + 15) / 16, H_, B_), 256, 0, stream>>>(
        qh, khp, vth, oh);
    proj_gemm <<<dim3((MTOT + 127) / 128, C_ / 128), 256, 0, stream>>>(
        oh, proj_w, proj_b, out);
}
